// mlp_eqprop_88871463289519
// MI455X (gfx1250) — matrix operand order repaired and checked by reading, not compiled, not run on hardware
//
#include <hip/hip_runtime.h>

typedef __attribute__((ext_vector_type(16))) __bf16 v16bf;
typedef __attribute__((ext_vector_type(8)))  float  v8f;

#define TSTEPS 30

// Workspace layout (bf16 element offsets). All rows 16B-aligned.
#define W0B_OFF   0                      // [16][512]  B for d0:  B[k][n]=W0[n][k] (n rows, 10 valid)
#define W0TB_OFF  8192                   // [512][32]  B for d1b: B[k][n]=W0[k][n], K padded 10->32
#define W1B_OFF   (W0TB_OFF + 16384)     // [512][512] B for d1a: B[k][n]=W1[n][k]
#define W1TB_OFF  (W1B_OFF + 262144)     // [512][512] B for d2b: B[k][n]=W1[k][n]
#define W2B_OFF   (W1TB_OFF + 262144)    // [512][800] B for d2a: B[k][n]=W2[n][k], K padded 784->800
#define WS_ELEMS  (W2B_OFF + 409600)

union FragU { uint4 u[2]; v16bf v; };
union PackU { __bf16 h[2]; unsigned int u; };

__device__ __forceinline__ v8f wmma_bf16(v16bf a, v16bf b, v8f c) {
  return __builtin_amdgcn_wmma_f32_16x16x32_bf16(false, a, false, b, (short)0, c, false, false);
}

// A fragment (16xK row-major source, one matrix per wave).
// ISA layout: lanes 0-15 hold row m=lane, K {0..7,16..23}; lanes 16-31 row m=lane-16, K {8..15,24..31}.
__device__ __forceinline__ v16bf ldA(const __bf16* base, int stride, int lane, int kt) {
  int m  = lane & 15;
  int k0 = kt + ((lane & 16) ? 8 : 0);
  const __bf16* p = base + m * stride + k0;
  FragU f;
  f.u[0] = *(const uint4*)(p);        // K k0..k0+7
  f.u[1] = *(const uint4*)(p + 16);   // K k0+16..k0+23
  return f.v;
}

// B fragment from row-major Bsrc[N][K] (so B[k][n] = Bsrc[n][k]).
// ISA layout: lanes 0-15 hold col n=lane, K {0..7,16..23}; lanes 16-31 col n=lane-16, K {8..15,24..31}.
__device__ __forceinline__ v16bf ldB(const __bf16* base, int stride, int lane, int nT, int kt) {
  int n  = nT + (lane & 15);
  int k0 = kt + ((lane & 16) ? 8 : 0);
  const __bf16* p = base + n * stride + k0;
  FragU f;
  f.u[0] = *(const uint4*)(p);        // K k0..k0+7
  f.u[1] = *(const uint4*)(p + 16);   // K k0+16..k0+23
  return f.v;
}

// Dual-N-tile GEMM inner loop. unroll 2 gives the scheduler two independent
// load/WMMA bodies to software-pipeline (no manual rotation -> no v_mov chains).
__device__ __forceinline__ void pair_gemm(const __bf16* A, int astride,
                                          const __bf16* B, int bstride,
                                          int nT0, int nT1, int lane, int iters,
                                          v8f& acc0, v8f& acc1) {
#pragma unroll 2
  for (int kt = 0; kt < iters; ++kt) {
    v16bf a = ldA(A, astride, lane, kt * 32);
    acc0 = wmma_bf16(a, ldB(B, bstride, lane, nT0, kt * 32), acc0);
    acc1 = wmma_bf16(a, ldB(B, bstride, lane, nT1, kt * 32), acc1);
  }
}

// Jacobi update: s = clip(0.5*s + 0.5*pre, 0, 1). C layout: lane=n (mod16), VGPR j -> m=j+8*(lane>=16).
__device__ __forceinline__ void upd_tile(float* s, int stride, int nT, int lane, v8f acc) {
  int n  = nT + (lane & 15);
  int mb = (lane & 16) ? 8 : 0;
#pragma unroll
  for (int j = 0; j < 8; ++j) {
    float o = s[(mb + j) * stride + n];
    float v = 0.5f * o + 0.5f * acc[j];
    v = fminf(fmaxf(v, 0.f), 1.f);
    s[(mb + j) * stride + n] = v;
  }
}

__global__ void __launch_bounds__(256)
prep_weights(const float* W0, const float* W1, const float* W2, __bf16* ws) {
  __bf16* W0b  = ws + W0B_OFF;
  __bf16* W0Tb = ws + W0TB_OFF;
  __bf16* W1b  = ws + W1B_OFF;
  __bf16* W1Tb = ws + W1TB_OFF;
  __bf16* W2b  = ws + W2B_OFF;
  for (int i = blockIdx.x * blockDim.x + threadIdx.x; i < WS_ELEMS;
       i += gridDim.x * blockDim.x) {
    if (i < W0TB_OFF) {                        // W0b [16][512], rows 10..15 zero
      int n = i >> 9, k = i & 511;
      W0b[i] = (n < 10) ? (__bf16)W0[n * 512 + k] : (__bf16)0.f;
    } else if (i < W1B_OFF) {                  // W0Tb [512][32], cols 10..31 zero
      int j = i - W0TB_OFF; int n = j >> 5, k = j & 31;
      W0Tb[j] = (k < 10) ? (__bf16)W0[k * 512 + n] : (__bf16)0.f;
    } else if (i < W1TB_OFF) {                 // W1b [512][512] direct
      int j = i - W1B_OFF;
      W1b[j] = (__bf16)W1[j];
    } else if (i < W2B_OFF) {                  // W1Tb [512][512] transposed
      int j = i - W1TB_OFF; int n = j >> 9, k = j & 511;
      W1Tb[j] = (__bf16)W1[k * 512 + n];
    } else {                                   // W2b [512][800], cols 784..799 zero
      int j = i - W2B_OFF; int n = j / 800, k = j - n * 800;
      W2b[j] = (k < 784) ? (__bf16)W2[n * 784 + k] : (__bf16)0.f;
    }
  }
}

__global__ void __launch_bounds__(256, 4)
eqprop_main(const float* s0_in, const float* s1_in, const float* s2_in,
            const float* x_in, const float* b0, const float* b1, const float* b2,
            const __bf16* ws, float* out) {
  __shared__ __align__(16) float  s0s[16][16];
  __shared__ __align__(16) float  s1s[16][512];
  __shared__ __align__(16) float  s2s[16][512];
  __shared__ __align__(16) __bf16 a0s[16][32];
  __shared__ __align__(16) __bf16 a1s[16][512];
  __shared__ __align__(16) __bf16 a2s[16][512];
  __shared__ __align__(16) __bf16 rxs[16][800];

  const int tid  = threadIdx.x;
  const int lane = tid & 31;
  const int wave = tid >> 5;
  const int rowBase = blockIdx.x * 16;

  const __bf16* W0b  = ws + W0B_OFF;
  const __bf16* W0Tb = ws + W0TB_OFF;
  const __bf16* W1b  = ws + W1B_OFF;
  const __bf16* W1Tb = ws + W1TB_OFF;
  const __bf16* W2b  = ws + W2B_OFF;

  // ---- init: pull 16-row batch tile into LDS ----
  {
    int m = tid >> 4, n = tid & 15;
    s0s[m][n] = (n < 10) ? s0_in[(rowBase + m) * 10 + n] : 0.f;
  }
  for (int i = tid; i < 16 * 512; i += 256) {
    (&s1s[0][0])[i] = s1_in[(rowBase + (i >> 9)) * 512 + (i & 511)];
    (&s2s[0][0])[i] = s2_in[(rowBase + (i >> 9)) * 512 + (i & 511)];
  }
  for (int i = tid; i < 16 * 32; i += 256) (&a0s[0][0])[i] = (__bf16)0.f;
  for (int i = tid; i < 16 * 800; i += 256) {
    int m = i / 800, k = i - m * 800;
    float v = (k < 784) ? x_in[(rowBase + m) * 784 + k] : 0.f;
    rxs[m][k] = (__bf16)fminf(fmaxf(v, 0.f), 1.f);
  }
  __syncthreads();

  // ---- 30 relaxation steps, fully on-chip ----
#pragma unroll 1
  for (int step = 0; step < TSTEPS; ++step) {
    // snapshot rho(s) -> bf16 activation buffers (2 elems/thread/iter, packed b32 stores)
    for (int i = tid * 2; i < 16 * 512; i += 512) {
      float2 v1 = *(const float2*)(&(&s1s[0][0])[i]);
      PackU p1;
      p1.h[0] = (__bf16)fminf(fmaxf(v1.x, 0.f), 1.f);
      p1.h[1] = (__bf16)fminf(fmaxf(v1.y, 0.f), 1.f);
      *(unsigned int*)(&(&a1s[0][0])[i]) = p1.u;
      float2 v2 = *(const float2*)(&(&s2s[0][0])[i]);
      PackU p2;
      p2.h[0] = (__bf16)fminf(fmaxf(v2.x, 0.f), 1.f);
      p2.h[1] = (__bf16)fminf(fmaxf(v2.y, 0.f), 1.f);
      *(unsigned int*)(&(&a2s[0][0])[i]) = p2.u;
    }
    if (tid < 160) {
      int m = tid / 10, n = tid - m * 10;
      a0s[m][n] = (__bf16)fminf(fmaxf(s0s[m][n], 0.f), 1.f);
    }
    __syncthreads();

    // d0 (16x10 output) handled entirely by wave 0: pre0 = rho(s1) @ W0^T + b0
    if (wave == 0) {
      float bv = ((lane & 15) < 10) ? b0[lane & 15] : 0.f;
      v8f acc = {bv, bv, bv, bv, bv, bv, bv, bv};
#pragma unroll 2
      for (int kt = 0; kt < 16; ++kt)
        acc = wmma_bf16(ldA(&a1s[0][0], 512, lane, kt * 32),
                        ldB(W0b, 512, lane, 0, kt * 32), acc);
      int n  = lane & 15;
      int mb = (lane & 16) ? 8 : 0;
      if (n < 10) {
#pragma unroll
        for (int j = 0; j < 8; ++j) {
          float o = s0s[mb + j][n];
          float v = fminf(fmaxf(0.5f * o + 0.5f * acc[j], 0.f), 1.f);
          s0s[mb + j][n] = v;
        }
      }
    }

    // waves 0-3: s1 tiles (N=512); waves 4-7: s2 tiles (N=512).
    // Each wave processes 8 N-tiles as 4 pairs: shared A fragment, two WMMA chains.
    if (wave < 4) {
#pragma unroll 1
      for (int p = 0; p < 4; ++p) {
        int nT0 = wave * 128 + p * 32;
        int nT1 = nT0 + 16;
        float bv0 = b1[nT0 + (lane & 15)];
        float bv1 = b1[nT1 + (lane & 15)];
        v8f acc0 = {bv0, bv0, bv0, bv0, bv0, bv0, bv0, bv0};
        v8f acc1 = {bv1, bv1, bv1, bv1, bv1, bv1, bv1, bv1};
        // pre1 = rho(s2) @ W1^T + ...
        pair_gemm(&a2s[0][0], 512, W1b, 512, nT0, nT1, lane, 16, acc0, acc1);
        // ... + rho(s0) @ W0 + b1
        {
          v16bf a = ldA(&a0s[0][0], 32, lane, 0);
          acc0 = wmma_bf16(a, ldB(W0Tb, 32, lane, nT0, 0), acc0);
          acc1 = wmma_bf16(a, ldB(W0Tb, 32, lane, nT1, 0), acc1);
        }
        upd_tile(&s1s[0][0], 512, nT0, lane, acc0);
        upd_tile(&s1s[0][0], 512, nT1, lane, acc1);
      }
    } else {
#pragma unroll 1
      for (int p = 0; p < 4; ++p) {
        int nT0 = (wave - 4) * 128 + p * 32;
        int nT1 = nT0 + 16;
        float bv0 = b2[nT0 + (lane & 15)];
        float bv1 = b2[nT1 + (lane & 15)];
        v8f acc0 = {bv0, bv0, bv0, bv0, bv0, bv0, bv0, bv0};
        v8f acc1 = {bv1, bv1, bv1, bv1, bv1, bv1, bv1, bv1};
        // pre2 = rho(x) @ W2^T + ...
        pair_gemm(&rxs[0][0], 800, W2b, 800, nT0, nT1, lane, 25, acc0, acc1);
        // ... + rho(s1) @ W1 + b2
        pair_gemm(&a1s[0][0], 512, W1Tb, 512, nT0, nT1, lane, 16, acc0, acc1);
        upd_tile(&s2s[0][0], 512, nT0, lane, acc0);
        upd_tile(&s2s[0][0], 512, nT1, lane, acc1);
      }
    }
    __syncthreads();
  }

  // ---- write concatenated [s0 | s1 | s2] = [B,1034] ----
  for (int i = tid; i < 16 * 1034; i += 256) {
    int m = i / 1034, c = i - m * 1034;
    float v;
    if (c < 10)       v = s0s[m][c];
    else if (c < 522) v = s1s[m][c - 10];
    else              v = s2s[m][c - 522];
    out[(rowBase + m) * 1034 + c] = v;
  }
}

extern "C" void kernel_launch(void* const* d_in, const int* in_sizes, int n_in,
                              void* d_out, int out_size, void* d_ws, size_t ws_size,
                              hipStream_t stream) {
  (void)in_sizes; (void)n_in; (void)out_size; (void)ws_size;
  const float* s0 = (const float*)d_in[0];
  const float* s1 = (const float*)d_in[1];
  const float* s2 = (const float*)d_in[2];
  const float* x  = (const float*)d_in[3];
  const float* W0 = (const float*)d_in[4];
  const float* b0 = (const float*)d_in[5];
  const float* W1 = (const float*)d_in[6];
  const float* b1 = (const float*)d_in[7];
  const float* W2 = (const float*)d_in[8];
  const float* b2 = (const float*)d_in[9];
  __bf16* ws = (__bf16*)d_ws;

  prep_weights<<<1024, 256, 0, stream>>>(W0, W1, W2, ws);
  eqprop_main<<<4096 / 16, 256, 0, stream>>>(s0, s1, s2, x, b0, b1, b2, ws,
                                             (float*)d_out);
}
